// SwinTransformerBlock_kv_29832842838661
// MI455X (gfx1250) — matrix operand order repaired and checked by reading, not compiled, not run on hardware
//
#include <hip/hip_runtime.h>
#include <hip/hip_bf16.h>
#include <math.h>

// ---------------- problem constants ----------------
#define BATCH   2
#define S_RES   16
#define H_RES   32
#define W_RES   32
#define L_TOK   (S_RES * H_RES * W_RES)          // 16384
#define CDIM    192
#define NHEAD   6
#define HD      32
#define NTOK    64                                // 4*4*4 window tokens
#define NWIN_B  256                               // windows per batch
#define BWIN    (BATCH * NWIN_B)                  // 512
#define MROWS   (BWIN * NTOK)                     // 32768 (== BATCH*L_TOK)
#define HID     (4 * CDIM)                        // 768
#define QSCALE  0.17677669529663687f              // 32^-0.5

typedef __attribute__((ext_vector_type(16))) _Float16 v16h;
typedef __attribute__((ext_vector_type(8)))  _Float16 v8h;
typedef __attribute__((ext_vector_type(8)))  float    v8f;
typedef __attribute__((ext_vector_type(4)))  int      v4i;

// ---------------- CDNA5 async global->LDS support (guarded) ----------------
#if defined(__has_builtin)
#if __has_builtin(__builtin_amdgcn_global_load_async_to_lds_b128)
#define HAS_ASYNC_LDS 1
#endif
#if __has_builtin(__builtin_amdgcn_s_wait_asynccnt)
#define HAS_WAIT_ASYNC 1
#endif
#endif
#ifndef HAS_ASYNC_LDS
#define HAS_ASYNC_LDS 0
#endif
#ifndef HAS_WAIT_ASYNC
#define HAS_WAIT_ASYNC 0
#endif

#if HAS_ASYNC_LDS
typedef __attribute__((address_space(1))) v4i* gv4i_ptr;
typedef __attribute__((address_space(3))) v4i* lv4i_ptr;

__device__ __forceinline__ void async_copy_b128(const void* g, void* l) {
  // generic->AS1 / generic->AS3 via integer round-trip (folded by InferAddressSpaces)
  __builtin_amdgcn_global_load_async_to_lds_b128(
      reinterpret_cast<gv4i_ptr>(reinterpret_cast<uintptr_t>(g)),
      reinterpret_cast<lv4i_ptr>(reinterpret_cast<uintptr_t>(l)),
      0, 0);
}
__device__ __forceinline__ void wait_async_le3() {
#if HAS_WAIT_ASYNC
  __builtin_amdgcn_s_wait_asynccnt(3);
#else
  asm volatile("s_wait_asynccnt 0x3" ::: "memory");
#endif
}
__device__ __forceinline__ void wait_async_le0() {
#if HAS_WAIT_ASYNC
  __builtin_amdgcn_s_wait_asynccnt(0);
#else
  asm volatile("s_wait_asynccnt 0x0" ::: "memory");
#endif
}
#endif

// -------------------------------------------------------------------------
// helpers
// -------------------------------------------------------------------------
__device__ __forceinline__ v16h pack16(v8h lo, v8h hi) {
  v16h r;
#pragma unroll
  for (int i = 0; i < 8; ++i) { r[i] = lo[i]; r[i + 8] = hi[i]; }
  return r;
}

__device__ __forceinline__ float block_reduce_sum(float v, float* sh, int c, int n) {
  sh[c] = v;
  __syncthreads();
  for (int s = 128; s > 0; s >>= 1) {
    if (c < s && c + s < n) sh[c] += sh[c + s];
    __syncthreads();
  }
  float r = sh[0];
  __syncthreads();
  return r;
}

// one K-step of the 8-wave 128x64 tile: 4 WMMAs per wave
__device__ __forceinline__ void gemm_step(const _Float16 (*__restrict__ As)[40],
                                          const _Float16 (*__restrict__ Bs)[40],
                                          int wr, int laneLo, int laneHi,
                                          v8f acc[4]) {
  // A frag (ISA 16-bit A 16x32): lane<16 -> K {0..7,16..23}; lane>=16 -> {8..15,24..31}
  const _Float16* arow = &As[wr + laneLo][0];
  v16h afrag = pack16(*(const v8h*)(arow + laneHi * 8),
                      *(const v8h*)(arow + laneHi * 8 + 16));
#pragma unroll
  for (int nj = 0; nj < 4; ++nj) {
    // B frag: col = 16*nj + laneLo, K matches A: lane<16 -> {0..7,16..23}; lane>=16 -> {8..15,24..31}
    const _Float16* brow = &Bs[nj * 16 + laneLo][laneHi * 8];
    v16h bfrag = pack16(*(const v8h*)(brow), *(const v8h*)(brow + 16));
    acc[nj] = __builtin_amdgcn_wmma_f32_16x16x32_f16(
        false, afrag, false, bfrag, (short)0, acc[nj], false, false);
  }
}

// -------------------------------------------------------------------------
// weight conversion: f32 [K][N] -> f16 N-major [N][K]
// -------------------------------------------------------------------------
__global__ void cvt_transpose_kernel(const float* __restrict__ src,
                                     _Float16* __restrict__ dst, int K, int N) {
  int i = blockIdx.x * blockDim.x + threadIdx.x;
  const int total = K * N;
  for (; i < total; i += gridDim.x * blockDim.x) {
    const int n = i / K, k = i - n * K;
    dst[i] = (_Float16)src[(size_t)k * N + n];
  }
}

// -------------------------------------------------------------------------
// expand relative-position-bias table -> dense [NHEAD][64][64]
// -------------------------------------------------------------------------
__global__ void build_bias_kernel(const float* __restrict__ rpb,
                                  float* __restrict__ bm) {
  const int i = blockIdx.x * blockDim.x + threadIdx.x;
  if (i >= NHEAD * NTOK * NTOK) return;
  const int head = i >> 12;
  const int r = i & 4095;
  const int t1 = r >> 6, t2 = r & 63;
  const int s1 = t1 >> 4, h1 = (t1 >> 2) & 3, w1 = t1 & 3;
  const int s2 = t2 >> 4, h2 = (t2 >> 2) & 3, w2 = t2 & 3;
  // replicates the reference's quirky index math
  const int idx = (s1 - s2 + 3) * 11 + (h1 - h2 + 3) * 7 + (w1 - w2 + 3);
  bm[i] = rpb[idx * NHEAD + head];
}

// -------------------------------------------------------------------------
// LayerNorm(skip) & LayerNorm(x_up) + 3D window partition -> f16
// -------------------------------------------------------------------------
__global__ __launch_bounds__(CDIM)
void ln_partition_kernel(const float* __restrict__ skip,
                         const float* __restrict__ xup,
                         const float* __restrict__ g,
                         const float* __restrict__ b,
                         _Float16* __restrict__ s_h,
                         _Float16* __restrict__ q_h) {
  __shared__ float sh[CDIM];
  const int row = blockIdx.x;
  const int c   = threadIdx.x;

  const float sv = skip[(size_t)row * CDIM + c];
  const float qv = xup[(size_t)row * CDIM + c];

  const float inv = 1.0f / CDIM;
  const float ms  = block_reduce_sum(sv, sh, c, CDIM) * inv;
  const float vs  = block_reduce_sum((sv - ms) * (sv - ms), sh, c, CDIM) * inv;
  const float mq  = block_reduce_sum(qv, sh, c, CDIM) * inv;
  const float vq  = block_reduce_sum((qv - mq) * (qv - mq), sh, c, CDIM) * inv;

  const float gn = g[c], bn = b[c];
  const float ns = (sv - ms) * rsqrtf(vs + 1e-5f) * gn + bn;
  const float nq = (qv - mq) * rsqrtf(vq + 1e-5f) * gn + bn;

  const int bb  = row >> 14;
  const int l   = row & (L_TOK - 1);
  const int s   = l >> 10;
  const int rem = l & 1023;
  const int h   = rem >> 5;
  const int w   = rem & 31;
  const int win = bb * NWIN_B + (s >> 2) * 64 + (h >> 2) * 8 + (w >> 2);
  const int t   = (s & 3) * 16 + (h & 3) * 4 + (w & 3);
  const size_t dst = ((size_t)win * NTOK + t) * CDIM + c;

  s_h[dst] = (_Float16)ns;
  q_h[dst] = (_Float16)(nq * QSCALE);
}

// -------------------------------------------------------------------------
// LayerNorm2 (row order preserved) -> f16
// -------------------------------------------------------------------------
__global__ __launch_bounds__(CDIM)
void ln2_kernel(const float* __restrict__ x,
                const float* __restrict__ g,
                const float* __restrict__ b,
                _Float16* __restrict__ out) {
  __shared__ float sh[CDIM];
  const int row = blockIdx.x;
  const int c   = threadIdx.x;
  const float v = x[(size_t)row * CDIM + c];
  const float inv = 1.0f / CDIM;
  const float m  = block_reduce_sum(v, sh, c, CDIM) * inv;
  const float vv = block_reduce_sum((v - m) * (v - m), sh, c, CDIM) * inv;
  out[(size_t)row * CDIM + c] =
      (_Float16)((v - m) * rsqrtf(vv + 1e-5f) * g[c] + b[c]);
}

// -------------------------------------------------------------------------
// Generic WMMA GEMM: C[M x N] = A[M x K] * Bt^T  (Bt stored N-major [N][K])
// block: 256 threads = 8 waves; tile 128(M) x 64(N); K-step 32
// async double-buffered global->LDS staging when available
// -------------------------------------------------------------------------
#define EPI_KV   0
#define EPI_PROJ 1
#define EPI_FC1  2
#define EPI_FC2  3

template <int EPI>
__global__ __launch_bounds__(256)
void wmma_gemm_kernel(const _Float16* __restrict__ A,
                      const _Float16* __restrict__ Bt,   // [Ncols][K]
                      const float* __restrict__ bias,
                      int M, int Ncols, int K,
                      _Float16* __restrict__ outH,   // KV: k  | FC1: hidden
                      _Float16* __restrict__ outH2,  // KV: v
                      float* __restrict__ outF,      // PROJ: x_res | FC2: d_out
                      const float* __restrict__ res) // PROJ: x | FC2: x_res
{
  const int tid    = threadIdx.x;
  const int waveId = tid >> 5;
  const int lane   = tid & 31;
  const int laneLo = lane & 15;
  const int laneHi = lane >> 4;
  const int mBase  = blockIdx.x * 128;
  const int nBase  = blockIdx.y * 64;
  const int wr     = waveId * 16;

  v8f acc[4];
#pragma unroll
  for (int nj = 0; nj < 4; ++nj) acc[nj] = {};

#if HAS_ASYNC_LDS
  __shared__ _Float16 As[2][128][40];   // 32 K + 8 pad (16B-aligned chunks)
  __shared__ _Float16 Bs[2][64][40];

  // per-thread chunk assignment (16B chunks, contiguous on both sides)
  const int ar0 = tid >> 2,         ac0 = (tid & 3) * 8;        // A chunks 0..255
  const int ar1 = (tid + 256) >> 2, ac1 = (tid & 3) * 8;        // A chunks 256..511
  const int bn_ = tid >> 2,         bc_ = (tid & 3) * 8;        // B chunks 0..255

  const int nk = K >> 5;
  // prologue: tile 0
  {
    async_copy_b128(A + (size_t)(mBase + ar0) * K + ac0, &As[0][ar0][ac0]);
    async_copy_b128(A + (size_t)(mBase + ar1) * K + ac1, &As[0][ar1][ac1]);
    async_copy_b128(Bt + (size_t)(nBase + bn_) * K + bc_, &Bs[0][bn_][bc_]);
  }
  for (int t = 0; t < nk; ++t) {
    const int kNext = (t + 1) << 5;
    if (t + 1 < nk) {   // issue next tile into the other buffer
      const int nb = (t + 1) & 1;
      async_copy_b128(A + (size_t)(mBase + ar0) * K + kNext + ac0, &As[nb][ar0][ac0]);
      async_copy_b128(A + (size_t)(mBase + ar1) * K + kNext + ac1, &As[nb][ar1][ac1]);
      async_copy_b128(Bt + (size_t)(nBase + bn_) * K + kNext + bc_, &Bs[nb][bn_][bc_]);
      wait_async_le3();           // tile t's 3 copies done; next 3 in flight
    } else {
      wait_async_le0();
    }
    __syncthreads();
    gemm_step(As[t & 1], Bs[t & 1], wr, laneLo, laneHi, acc);
    __syncthreads();              // all waves done reading before overwrite
  }
#else
  __shared__ _Float16 As1[128][40];
  __shared__ _Float16 Bs1[64][40];
  const int ar = tid >> 1;             // A staging: row 0..127, 16 halves each
  const int ac = (tid & 1) * 16;
  const int bn_ = tid >> 2;            // B staging: row 0..63, 8 halves each
  const int bc_ = (tid & 3) * 8;

  for (int kBase = 0; kBase < K; kBase += 32) {
    __syncthreads();
    const _Float16* ap = A + (size_t)(mBase + ar) * K + kBase + ac;
    *(v8h*)(&As1[ar][ac])     = *(const v8h*)(ap);
    *(v8h*)(&As1[ar][ac + 8]) = *(const v8h*)(ap + 8);
    if (kBase + 32 < K) __builtin_prefetch(ap + 32, 0, 0);
    *(v8h*)(&Bs1[bn_][bc_]) =
        *(const v8h*)(Bt + (size_t)(nBase + bn_) * K + kBase + bc_);
    __syncthreads();
    gemm_step(As1, Bs1, wr, laneLo, laneHi, acc);
  }
#endif

  // ---- epilogue ----
#pragma unroll
  for (int nj = 0; nj < 4; ++nj) {
    const int col = nBase + nj * 16 + laneLo;
    const float bcol = bias[col];
#pragma unroll
    for (int r = 0; r < 8; ++r) {
      const int row = mBase + wr + r + laneHi * 8;   // D layout: M = r + 8*laneHi
      float v = acc[nj][r] + bcol;

      if (EPI == EPI_KV) {
        if (col < CDIM) outH [(size_t)row * CDIM + col]          = (_Float16)v;
        else            outH2[(size_t)row * CDIM + (col - CDIM)] = (_Float16)v;
      } else if (EPI == EPI_PROJ) {
        const int win = row >> 6, t = row & 63;
        const int bb  = win >> 8, wi = win & 255;
        const int s   = ((wi >> 6) << 2)       + (t >> 4);
        const int hh  = (((wi >> 3) & 7) << 2) + ((t >> 2) & 3);
        const int ww  = ((wi & 7) << 2)        + (t & 3);
        const size_t dst =
            ((size_t)bb * L_TOK + (s << 10) + (hh << 5) + ww) * CDIM + col;
        outF[dst] = v + res[dst];
      } else if (EPI == EPI_FC1) {
        const float gv = 0.5f * v * (1.0f + erff(v * 0.70710678118654752f));
        outH[(size_t)row * HID + col] = (_Float16)gv;
      } else { // EPI_FC2
        const size_t dst = (size_t)row * CDIM + col;
        outF[dst] = v + res[dst];
      }
    }
  }
}

// -------------------------------------------------------------------------
// Window attention: one wave per (window, head)
// -------------------------------------------------------------------------
__global__ __launch_bounds__(32)
void attn_kernel(const _Float16* __restrict__ q_h,
                 const _Float16* __restrict__ k_h,
                 const _Float16* __restrict__ v_h,
                 const float* __restrict__ bm,      // [NHEAD][64][64] dense bias
                 _Float16* __restrict__ ao_h) {
  __shared__ _Float16 Pt[64][72];   // probs, padded (row stride 144B)
  __shared__ _Float16 Vt[32][72];   // v transposed: [feat][token]

  const int wg   = blockIdx.x;
  const int win  = wg / NHEAD;
  const int head = wg % NHEAD;
  const int lane   = threadIdx.x;
  const int laneLo = lane & 15;
  const int laneHi = lane >> 4;

  const size_t wbase = (size_t)win * NTOK * CDIM + head * HD;
  const float* bmh   = bm + head * (NTOK * NTOK);

  // ---- stage V transposed into LDS ----
  for (int tok = lane; tok < NTOK; tok += 32) {
    const _Float16* vp = v_h + wbase + (size_t)tok * CDIM;
#pragma unroll
    for (int f0 = 0; f0 < HD; f0 += 8) {
      v8h tv = *(const v8h*)(vp + f0);
#pragma unroll
      for (int i = 0; i < 8; ++i) Vt[f0 + i][tok] = tv[i];
    }
  }

  // ---- k fragments ----
  v16h bk[4];
#pragma unroll
  for (int nj = 0; nj < 4; ++nj) {
    const _Float16* kp =
        k_h + wbase + (size_t)(nj * 16 + laneLo) * CDIM + laneHi * 8;
    bk[nj] = pack16(*(const v8h*)(kp), *(const v8h*)(kp + 16));
  }

  // ---- scores + bias + softmax, one 16-row band at a time ----
  for (int mi = 0; mi < 4; ++mi) {
    const _Float16* qp =
        q_h + wbase + (size_t)(mi * 16 + laneLo) * CDIM + laneHi * 8;
    v16h aq = pack16(*(const v8h*)(qp), *(const v8h*)(qp + 16));

    v8f sc[4];
#pragma unroll
    for (int nj = 0; nj < 4; ++nj) {
      v8f z = {};
      sc[nj] = __builtin_amdgcn_wmma_f32_16x16x32_f16(
          false, aq, false, bk[nj], (short)0, z, false, false);
    }

#pragma unroll
    for (int r = 0; r < 8; ++r) {
      const int t1 = mi * 16 + r + laneHi * 8;
      const float* brow = bmh + t1 * NTOK + laneLo;   // coalesced across lanes
      float rowv[4];
#pragma unroll
      for (int nj = 0; nj < 4; ++nj) rowv[nj] = sc[nj][r] + brow[nj * 16];

      float m = fmaxf(fmaxf(rowv[0], rowv[1]), fmaxf(rowv[2], rowv[3]));
#pragma unroll
      for (int d = 8; d >= 1; d >>= 1) m = fmaxf(m, __shfl_xor(m, d, 32));
      float ssum = 0.f;
#pragma unroll
      for (int nj = 0; nj < 4; ++nj) {
        rowv[nj] = __expf(rowv[nj] - m);
        ssum += rowv[nj];
      }
#pragma unroll
      for (int d = 8; d >= 1; d >>= 1) ssum += __shfl_xor(ssum, d, 32);
      const float rinv = 1.0f / ssum;
#pragma unroll
      for (int nj = 0; nj < 4; ++nj)
        Pt[t1][nj * 16 + laneLo] = (_Float16)(rowv[nj] * rinv);
    }
  }
  __syncthreads();

  // ---- out = P @ V ----
  v8f oc[4][2];
#pragma unroll
  for (int mi = 0; mi < 4; ++mi)
#pragma unroll
    for (int nj = 0; nj < 2; ++nj) oc[mi][nj] = {};

#pragma unroll
  for (int kk = 0; kk < 2; ++kk) {
    v16h ap[4], bv[2];
#pragma unroll
    for (int mi = 0; mi < 4; ++mi) {
      const _Float16* pp = &Pt[mi * 16 + laneLo][kk * 32 + laneHi * 8];
      ap[mi] = pack16(*(const v8h*)(pp), *(const v8h*)(pp + 16));
    }
#pragma unroll
    for (int nj = 0; nj < 2; ++nj) {
      const _Float16* vp = &Vt[nj * 16 + laneLo][kk * 32 + laneHi * 8];
      bv[nj] = pack16(*(const v8h*)(vp), *(const v8h*)(vp + 16));
    }
#pragma unroll
    for (int mi = 0; mi < 4; ++mi)
#pragma unroll
      for (int nj = 0; nj < 2; ++nj)
        oc[mi][nj] = __builtin_amdgcn_wmma_f32_16x16x32_f16(
            false, ap[mi], false, bv[nj], (short)0, oc[mi][nj], false, false);
  }

#pragma unroll
  for (int mi = 0; mi < 4; ++mi)
#pragma unroll
    for (int nj = 0; nj < 2; ++nj)
#pragma unroll
      for (int r = 0; r < 8; ++r) {
        const int t1  = mi * 16 + r + laneHi * 8;
        const int col = nj * 16 + laneLo;
        ao_h[wbase + (size_t)t1 * CDIM + col] = (_Float16)oc[mi][nj][r];
      }
}

// -------------------------------------------------------------------------
// launcher
// -------------------------------------------------------------------------
extern "C" void kernel_launch(void* const* d_in, const int* in_sizes, int n_in,
                              void* d_out, int out_size, void* d_ws, size_t ws_size,
                              hipStream_t stream) {
  const float* x      = (const float*)d_in[0];
  const float* skip   = (const float*)d_in[2];
  const float* x_up   = (const float*)d_in[3];
  const float* n1g    = (const float*)d_in[4];
  const float* n1b    = (const float*)d_in[5];
  const float* kv_w   = (const float*)d_in[6];
  const float* kv_b   = (const float*)d_in[7];
  const float* rpb    = (const float*)d_in[8];
  const float* proj_w = (const float*)d_in[9];
  const float* proj_b = (const float*)d_in[10];
  const float* n2g    = (const float*)d_in[11];
  const float* n2b    = (const float*)d_in[12];
  const float* fc1_w  = (const float*)d_in[13];
  const float* fc1_b  = (const float*)d_in[14];
  const float* fc2_w  = (const float*)d_in[15];
  const float* fc2_b  = (const float*)d_in[16];
  float* out = (float*)d_out;

  char* ws = (char*)d_ws;
  size_t off = 0;
  auto alloc = [&](size_t bytes) -> char* {
    char* p = ws + off;
    off += (bytes + 255) & ~(size_t)255;
    return p;
  };

  const size_t NE = (size_t)MROWS * CDIM;
  _Float16* s_h   = (_Float16*)alloc(NE * 2);
  _Float16* q_h   = (_Float16*)alloc(NE * 2);
  _Float16* k_h   = (_Float16*)alloc(NE * 2);
  _Float16* v_h   = (_Float16*)alloc(NE * 2);
  _Float16* ao_h  = (_Float16*)alloc(NE * 2);
  float*    x_res = (float*)   alloc(NE * 4);
  _Float16* hn_h  = (_Float16*)alloc(NE * 2);
  _Float16* hm_h  = (_Float16*)alloc((size_t)MROWS * HID * 2);
  _Float16* kvwT  = (_Float16*)alloc((size_t)(2 * CDIM) * CDIM * 2); // [384][192]
  _Float16* projwT= (_Float16*)alloc((size_t)CDIM * CDIM * 2);       // [192][192]
  _Float16* fc1wT = (_Float16*)alloc((size_t)HID * CDIM * 2);        // [768][192]
  _Float16* fc2wT = (_Float16*)alloc((size_t)CDIM * HID * 2);        // [192][768]
  float*    bm    = (float*)   alloc((size_t)NHEAD * NTOK * NTOK * 4);

  // 1) weight transpose+convert (f32 [K][N] -> f16 [N][K]) and bias expansion
  cvt_transpose_kernel<<<288, 256, 0, stream>>>(kv_w,  kvwT,  CDIM, 2 * CDIM);
  cvt_transpose_kernel<<<144, 256, 0, stream>>>(proj_w, projwT, CDIM, CDIM);
  cvt_transpose_kernel<<<576, 256, 0, stream>>>(fc1_w, fc1wT, CDIM, HID);
  cvt_transpose_kernel<<<576, 256, 0, stream>>>(fc2_w, fc2wT, HID, CDIM);
  build_bias_kernel<<<(NHEAD * NTOK * NTOK + 255) / 256, 256, 0, stream>>>(rpb, bm);

  // 2) LN + window partition
  ln_partition_kernel<<<MROWS, CDIM, 0, stream>>>(skip, x_up, n1g, n1b, s_h, q_h);

  // 3) kv GEMM
  wmma_gemm_kernel<EPI_KV><<<dim3(MROWS / 128, 384 / 64), 256, 0, stream>>>(
      s_h, kvwT, kv_b, MROWS, 2 * CDIM, CDIM, k_h, v_h, nullptr, nullptr);

  // 4) window attention
  attn_kernel<<<BWIN * NHEAD, 32, 0, stream>>>(q_h, k_h, v_h, bm, ao_h);

  // 5) proj GEMM + window reverse + residual
  wmma_gemm_kernel<EPI_PROJ><<<dim3(MROWS / 128, CDIM / 64), 256, 0, stream>>>(
      ao_h, projwT, proj_b, MROWS, CDIM, CDIM, nullptr, nullptr, x_res, x);

  // 6) LN2
  ln2_kernel<<<MROWS, CDIM, 0, stream>>>(x_res, n2g, n2b, hn_h);

  // 7) fc1 + GELU
  wmma_gemm_kernel<EPI_FC1><<<dim3(MROWS / 128, HID / 64), 256, 0, stream>>>(
      hn_h, fc1wT, fc1_b, MROWS, HID, CDIM, hm_h, nullptr, nullptr, nullptr);

  // 8) fc2 + residual -> d_out
  wmma_gemm_kernel<EPI_FC2><<<dim3(MROWS / 128, CDIM / 64), 256, 0, stream>>>(
      hm_h, fc2wT, fc2_b, MROWS, CDIM, HID, nullptr, nullptr, out, x_res);

  (void)in_sizes; (void)n_in; (void)out_size; (void)ws_size;
}
